// SelConv_43516608643456
// MI455X (gfx1250) — compile-verified, matrix operand order checked by reading, not run on hardware
//
#include <hip/hip_runtime.h>
#include <hip/hip_bf16.h>

// ---------------------------------------------------------------------------
// SelectionConv on MI455X (gfx1250):
//   pass s:  scatter  g[dst] += interp*x[src]   (edges with sel==s, f32 atomics)
//            gemm     out   += g @ W[s]         (bf16x3-split WMMA, TDM->LDS)
// Workspace: g (N*128 f32, 51.2MB, L2-resident) + packed W (hi/lo bf16).
// ---------------------------------------------------------------------------

typedef __bf16  v16bf __attribute__((ext_vector_type(16)));
typedef float   v8f   __attribute__((ext_vector_type(8)));
typedef unsigned int u32x4 __attribute__((ext_vector_type(4)));
typedef int          i32x4 __attribute__((ext_vector_type(4)));
typedef int          i32x8 __attribute__((ext_vector_type(8)));

#if __has_builtin(__builtin_amdgcn_tensor_load_to_lds) && __has_builtin(__builtin_amdgcn_s_wait_tensorcnt)
#define HAVE_TDM 1
#else
#define HAVE_TDM 0
#endif

union Frag {
    v16bf v;
    unsigned int u[8];
    uint4 q[2];
};

__device__ __forceinline__ unsigned rfl(unsigned v) {
    return (unsigned)__builtin_amdgcn_readfirstlane((int)v);
}

// hi part of bf16 split: truncate mantissa (exact residual captured by lo)
__device__ __forceinline__ unsigned pack_hi2(float a, float b) {
    return (__float_as_uint(a) >> 16) | (__float_as_uint(b) & 0xFFFF0000u);
}
__device__ __forceinline__ unsigned pack_lo2(float a, float b) {
    float ra = a - __uint_as_float(__float_as_uint(a) & 0xFFFF0000u);
    float rb = b - __uint_as_float(__float_as_uint(b) & 0xFFFF0000u);
    return (__float_as_uint(ra) >> 16) | (__float_as_uint(rb) & 0xFFFF0000u);
}
// packed u32: [15:0]=hi bf16, [31:16]=lo bf16
__device__ __forceinline__ unsigned pack_hl(float f) {
    unsigned u  = __float_as_uint(f);
    float    r  = f - __uint_as_float(u & 0xFFFF0000u);
    return (u >> 16) | (__float_as_uint(r) & 0xFFFF0000u);
}

// ---------------------------------------------------------------------------
// Kernel 0: pre-split W into packed hi/lo bf16 (once per launch)
// ---------------------------------------------------------------------------
__global__ __launch_bounds__(256) void prep_w_kernel(const float* __restrict__ w,
                                                     unsigned* __restrict__ w2, int n) {
    int i = blockIdx.x * 256 + threadIdx.x;
    if (i < n) w2[i] = pack_hl(w[i]);
}

// ---------------------------------------------------------------------------
// Kernel 1: edge scatter for one selection. One wave per edge.
//   lane handles 4 channels: g[dst*128 + lane*4 + 0..3] += interp * x[src,...]
// ---------------------------------------------------------------------------
__global__ __launch_bounds__(256) void scatter_kernel(const float* __restrict__ x,
                                                      const int* __restrict__ src,
                                                      const int* __restrict__ dst,
                                                      const int* __restrict__ selv,
                                                      const float* __restrict__ interp,
                                                      float* __restrict__ g,
                                                      int s, int E) {
    int e = __builtin_amdgcn_readfirstlane(
        (int)((blockIdx.x * blockDim.x + threadIdx.x) >> 5));
    if (e >= E) return;
    if (selv[e] != s) return;                 // uniform scalar branch: 8/9 waves exit
    int   sn = src[e];
    int   dn = dst[e];
    float wI = interp[e];
    int   lane = threadIdx.x & 31;
    float4 v = ((const float4*)(x + (size_t)sn * 128))[lane];   // 512B/row, coalesced
    float* gd = g + (size_t)dn * 128 + lane * 4;
#if defined(__HIP_DEVICE_COMPILE__)
    unsafeAtomicAdd(gd + 0, v.x * wI);
    unsafeAtomicAdd(gd + 1, v.y * wI);
    unsafeAtomicAdd(gd + 2, v.z * wI);
    unsafeAtomicAdd(gd + 3, v.w * wI);
#endif
}

// ---------------------------------------------------------------------------
// TDM: load one 16x128 f32 tile of g into LDS, padding rows 128->132 dwords
// (pad_interval=128 dwords -> code 6, pad_amount=4 dwords -> code 3).
// Toolchain exposes the 6-arg builtin: (g0:v4u, g1:v8i, g2:v4i, g3:v4i, v8i, cpol)
// groups 2/3 + trailing group are unused for a 2D tile -> zeros.
// ---------------------------------------------------------------------------
#if HAVE_TDM
__device__ __forceinline__ void tdm_load_tile(const float* gtile, unsigned lds_off) {
    unsigned long long ga = (unsigned long long)gtile;
    u32x4 q0;
    q0[0] = rfl(1u);                                   // count=1, user mode
    q0[1] = rfl(lds_off);                              // LDS byte address
    q0[2] = rfl((unsigned)(ga & 0xFFFFFFFFull));       // global_addr[31:0]
    q0[3] = rfl((unsigned)((ga >> 32) & 0x1FFFFFFu) | (2u << 30)); // addr[56:32] | type=2
    i32x8 q1;
    q1[0] = (int)rfl((2u << 16) | (1u << 20) | (6u << 22) | (3u << 25)); // 4B, pad on
    q1[1] = (int)rfl(128u << 16);   // tensor_dim0 = 128
    q1[2] = (int)rfl(16u << 16);    // tensor_dim1 = 16
    q1[3] = (int)rfl(128u << 16);   // tile_dim0 = 128
    q1[4] = (int)rfl(16u);          // tile_dim1 = 16, tile_dim2 = 0
    q1[5] = (int)rfl(128u);         // tensor_dim0_stride = 128
    q1[6] = 0;
    q1[7] = 0;
    i32x4 q2 = {0, 0, 0, 0};
    i32x4 q3 = {0, 0, 0, 0};
    i32x8 q4 = {0, 0, 0, 0, 0, 0, 0, 0};
    __builtin_amdgcn_tensor_load_to_lds(q0, q1, q2, q3, q4, 0);
}
#endif

// ---------------------------------------------------------------------------
// Kernel 2: out[t*16.., :] (+)= g_tile @ W[s]  with bf16x3 split.
//   8 waves/WG: wave w owns output columns [16w,16w+16).
//   B (W columns) built once per WG into 64 persistent VGPRs/wave.
//   A tiles: TDM -> LDS f32 (double buffered) -> packed hi/lo bf16 pairs in LDS.
//   After consuming a tile, its g region is re-zeroed for the next pass.
// ---------------------------------------------------------------------------
__global__ __launch_bounds__(256) void gemm_kernel(float* __restrict__ g,
                                                   const unsigned* __restrict__ w2,
                                                   const float* __restrict__ bias,
                                                   float* __restrict__ out,
                                                   int tiles, int accumulate) {
    __shared__ float    fbuf[2][16 * 132];   // f32 tile, padded rows (TDM dest)
    __shared__ unsigned Ahi[16 * 68];        // packed hi bf16 pairs, padded rows
    __shared__ unsigned Alo[16 * 68];        // packed lo bf16 pairs

    const int tid  = threadIdx.x;
    const int lane = tid & 31;
    const int wid  = tid >> 5;
    const int m16  = lane & 15;
    const int hsel = lane >> 4;              // 0: K 0-7/16-23, 1: K 8-15/24-31

    // ---- B fragments (persistent, once per workgroup) ----
    Frag Bh[4], Bl[4];
    {
        const int n     = (wid << 4) + m16;          // output column
        const int base0 = hsel ? 8 : 0;
#pragma unroll
        for (int k4 = 0; k4 < 4; ++k4) {
#pragma unroll
            for (int j = 0; j < 8; ++j) {
                int k0 = 32 * k4 + base0 + ((j < 4) ? (2 * j) : (16 + 2 * (j - 4)));
                unsigned a = w2[k0 * 128 + n];
                unsigned b = w2[(k0 + 1) * 128 + n];
                Bh[k4].u[j] = (a & 0xFFFFu) | (b << 16);
                Bl[k4].u[j] = (a >> 16) | (b & 0xFFFF0000u);
            }
        }
    }

    const int t0 = (int)blockIdx.x;
    const int G  = (int)gridDim.x;
    const int cnt = (tiles - t0 + G - 1) / G;        // uniform per WG
    if (cnt <= 0) return;

    const bool leader = (__builtin_amdgcn_readfirstlane((int)tid) < 32); // wave0

#if HAVE_TDM
    if (leader) tdm_load_tile(g + (size_t)t0 * 2048, (unsigned)(unsigned long long)&fbuf[0][0]);
#endif

    for (int i = 0; i < cnt; ++i) {
        const int t  = t0 + i * G;
        const int ib = i & 1;

#if HAVE_TDM
        if (leader) {
            if (i + 1 < cnt) {   // issue next tile, then wait for current
                tdm_load_tile(g + (size_t)(t0 + (i + 1) * G) * 2048,
                              (unsigned)(unsigned long long)&fbuf[ib ^ 1][0]);
                __builtin_amdgcn_s_wait_tensorcnt(1);
            } else {
                __builtin_amdgcn_s_wait_tensorcnt(0);
            }
        }
        __syncthreads();         // fbuf[ib] visible to all waves
#else
        {   // fallback: cooperative copy with row padding
            const float* gt = g + (size_t)t * 2048;
            int cm = tid >> 4, kb = (tid & 15) * 8;
            float4 a = ((const float4*)(gt + cm * 128 + kb))[0];
            float4 b = ((const float4*)(gt + cm * 128 + kb))[1];
            *(float4*)&fbuf[ib][cm * 132 + kb]     = a;
            *(float4*)&fbuf[ib][cm * 132 + kb + 4] = b;
        }
        __syncthreads();
#endif

        // ---- convert f32 tile -> packed hi/lo bf16 pairs (cooperative) ----
        {
            int cm = tid >> 4;                 // row 0..15
            int kb = (tid & 15) * 8;           // 8 consecutive K per thread
            const float* fr = &fbuf[ib][cm * 132 + kb];
            float4 f0 = *(const float4*)fr;
            float4 f1 = *(const float4*)(fr + 4);
            uint4 hv, lv;
            hv.x = pack_hi2(f0.x, f0.y);  hv.y = pack_hi2(f0.z, f0.w);
            hv.z = pack_hi2(f1.x, f1.y);  hv.w = pack_hi2(f1.z, f1.w);
            lv.x = pack_lo2(f0.x, f0.y);  lv.y = pack_lo2(f0.z, f0.w);
            lv.z = pack_lo2(f1.x, f1.y);  lv.w = pack_lo2(f1.z, f1.w);
            *(uint4*)&Ahi[cm * 68 + (kb >> 1)] = hv;
            *(uint4*)&Alo[cm * 68 + (kb >> 1)] = lv;
        }
        __syncthreads();

        // ---- re-zero this g tile (TDM already consumed it) for next pass ----
        {
            float4* z = (float4*)(g + (size_t)t * 2048);
            float4 z4 = {0.f, 0.f, 0.f, 0.f};
            z[tid]       = z4;
            z[tid + 256] = z4;
        }

        // ---- WMMA: C += Ah*Bh + Ah*Bl + Al*Bh over 4 K-chunks of 32 ----
        Frag ah, al;
        v8f  c = {};
#pragma unroll
        for (int k4 = 0; k4 < 4; ++k4) {
            const int p0 = 16 * k4 + (hsel ? 4 : 0);
            const uint4* hp = (const uint4*)&Ahi[m16 * 68 + p0];
            const uint4* lp = (const uint4*)&Alo[m16 * 68 + p0];
            ah.q[0] = hp[0];  ah.q[1] = hp[2];
            al.q[0] = lp[0];  al.q[1] = lp[2];
            c = __builtin_amdgcn_wmma_f32_16x16x32_bf16(false, ah.v, false, Bh[k4].v,
                                                        (short)0, c, false, false);
            c = __builtin_amdgcn_wmma_f32_16x16x32_bf16(false, ah.v, false, Bl[k4].v,
                                                        (short)0, c, false, false);
            c = __builtin_amdgcn_wmma_f32_16x16x32_bf16(false, al.v, false, Bh[k4].v,
                                                        (short)0, c, false, false);
        }

        // ---- epilogue: C layout -> out ----
        {
            const int    col     = (wid << 4) + m16;
            const size_t rowbase = (size_t)t * 16 + hsel * 8;
            if (accumulate) {
#pragma unroll
                for (int r = 0; r < 8; ++r) {
                    size_t o = (rowbase + r) * 128 + col;
                    out[o] = out[o] + c[r];
                }
            } else {
                float bv = bias[col];
#pragma unroll
                for (int r = 0; r < 8; ++r) {
                    size_t o = (rowbase + r) * 128 + col;
                    out[o] = c[r] + bv;
                }
            }
        }
        __syncthreads();   // fbuf[ib] free for the TDM issued 2 iters ahead
    }
}

// ---------------------------------------------------------------------------
// Host-side launcher (graph-capture safe: only kernels + hipMemsetAsync)
// ---------------------------------------------------------------------------
extern "C" void kernel_launch(void* const* d_in, const int* in_sizes, int n_in,
                              void* d_out, int out_size, void* d_ws, size_t ws_size,
                              hipStream_t stream) {
    const float* x      = (const float*)d_in[0];
    const int*   eidx   = (const int*)d_in[1];
    const int*   selv   = (const int*)d_in[2];
    const float* interp = (const float*)d_in[3];
    const float* weight = (const float*)d_in[4];
    const float* bias   = (const float*)d_in[5];
    float*       out    = (float*)d_out;

    const int N = in_sizes[0] / 128;
    const int E = in_sizes[1] / 2;
    const int S = in_sizes[4] / (128 * 128);
    const int* src = eidx;
    const int* dst = eidx + E;

    float* g = (float*)d_ws;
    size_t gbytes = (size_t)N * 128 * sizeof(float);
    unsigned* w2 = (unsigned*)((char*)d_ws + ((gbytes + 255) & ~(size_t)255));

    (void)hipMemsetAsync(g, 0, gbytes, stream);                 // pass 0 needs zeros
    prep_w_kernel<<<(S * 16384 + 255) / 256, 256, 0, stream>>>(weight, w2, S * 16384);

    const int tiles = N / 16;                                   // N = 100000 -> 6250
    const int scatter_blocks = (E + 7) / 8;                     // 1 wave per edge
    int gemm_blocks = 1024;
    if (gemm_blocks > tiles) gemm_blocks = tiles;

    for (int s = 0; s < S; ++s) {
        scatter_kernel<<<scatter_blocks, 256, 0, stream>>>(x, src, dst, selv, interp,
                                                           g, s, E);
        gemm_kernel<<<gemm_blocks, 256, 0, stream>>>(g, w2 + s * 128 * 128, bias, out,
                                                     tiles, s != 0);
        // gemm_kernel re-zeroes g, so the next scatter accumulates onto zeros and
        // the workspace is left all-zero after the final pass (replay-safe).
    }
}
